// GAT2_47467978555680
// MI455X (gfx1250) — matrix operand order repaired and checked by reading, not compiled, not run on hardware
//
#include <hip/hip_runtime.h>
#include <hip/hip_bf16.h>

// ---------------------------------------------------------------------------
// GATv2 x2 for MI455X (gfx1250): bf16 WMMA GEMMs + atomic edge softmax/scatter
// ---------------------------------------------------------------------------

#define NNODES 20000
#define NEDGES 320000
#define F_IN   256
#define HID    64
#define HEADS  8
#define NCLS   64
#define HC     (HEADS * HID)   // 512
#define NEG_SLOPE 0.2f
#define EPS 1e-16f

typedef __attribute__((ext_vector_type(16))) __bf16 v16bf;
typedef __attribute__((ext_vector_type(4)))  __bf16 v4bf;
typedef __attribute__((ext_vector_type(8)))  float  v8f;

// order-preserving float <-> uint key (for atomicMax over signed floats)
__device__ __forceinline__ unsigned fkey(float f) {
    unsigned u = __float_as_uint(f);
    return (u & 0x80000000u) ? ~u : (u | 0x80000000u);
}
__device__ __forceinline__ float funkey(unsigned k) {
    unsigned u = (k & 0x80000000u) ? (k & 0x7fffffffu) : ~k;
    return __uint_as_float(u);
}

__device__ __forceinline__ float leaky(float v) {
    return v > 0.f ? v : NEG_SLOPE * v;
}

// ---------------------------------------------------------------------------
// Tiled GEMM: C[M,N] = A[M,K] * B[K,N]  (fp32 in/out, bf16 WMMA compute)
// Block tile 64x64, K-tile 64, 256 threads = 8 waves; wave (wm,wn) owns 16x32.
// Requires: N % 64 == 0, K % 64 == 0 (M may be ragged).
// ---------------------------------------------------------------------------
#define TK 64
#define LPAD 8
__global__ __launch_bounds__(256)
void gemm_bf16_wmma(const float* __restrict__ A, const float* __restrict__ B,
                    float* __restrict__ C, int M, int N, int K) {
    __shared__ __bf16 Alds[64][TK + LPAD];  // [m][k] row-major (stride 144B)
    __shared__ __bf16 Blds[64][TK + LPAD];  // [n][k] transposed

    const int tid  = threadIdx.x;
    const int lane = tid & 31;
    const int wave = tid >> 5;
    const int wm   = wave & 3;    // 0..3 : M sub-tile * 16
    const int wn   = wave >> 2;   // 0..1 : N sub-tile * 32
    const int half = lane >> 4;
    const int l15  = lane & 15;

    const int m0 = blockIdx.x * 64;
    const int n0 = blockIdx.y * 64;
    const bool full = (m0 + 64 <= M);   // block-uniform fast path

    v8f acc0 = {}; v8f acc1 = {};

    for (int k0 = 0; k0 < K; k0 += TK) {
        // ---- stage A tile (64 x 64 floats -> bf16), 4 float4 per thread ----
        if (full) {
            #pragma unroll
            for (int i = 0; i < 4; ++i) {
                int idx = tid + i * 256;                 // 0..1023 float4 slots
                int r = idx >> 4, c = (idx & 15) << 2;   // row, col (mult of 4)
                const float4 f = *(const float4*)(A + (size_t)(m0 + r) * K + k0 + c);
                v4bf p = { (__bf16)f.x, (__bf16)f.y, (__bf16)f.z, (__bf16)f.w };
                *(v4bf*)&Alds[r][c] = p;                 // 8B LDS store, aligned
            }
        } else {
            #pragma unroll
            for (int i = 0; i < 4; ++i) {
                int idx = tid + i * 256;
                int r = idx >> 4, c = (idx & 15) << 2;
                float4 f = make_float4(0.f, 0.f, 0.f, 0.f);
                if (m0 + r < M)                          // one guard per float4
                    f = *(const float4*)(A + (size_t)(m0 + r) * K + k0 + c);
                v4bf p = { (__bf16)f.x, (__bf16)f.y, (__bf16)f.z, (__bf16)f.w };
                *(v4bf*)&Alds[r][c] = p;
            }
        }
        // ---- stage B tile (64k x 64n floats), transpose into [n][k] ----
        #pragma unroll
        for (int i = 0; i < 4; ++i) {
            int idx = tid + i * 256;
            int r = idx >> 4, c = (idx & 15) << 2;       // r = k, c = n
            const float4 f = *(const float4*)(B + (size_t)(k0 + r) * N + n0 + c);
            Blds[c + 0][r] = (__bf16)f.x;
            Blds[c + 1][r] = (__bf16)f.y;
            Blds[c + 2][r] = (__bf16)f.z;
            Blds[c + 3][r] = (__bf16)f.w;
        }
        __syncthreads();

        // ---- 2 K-halves x 2 N-subtiles = 4 WMMA per wave per tile ----
        #pragma unroll
        for (int kh = 0; kh < 2; ++kh) {
            const int kb = kh * 32;
            union { unsigned u[8]; v16bf v; } af, bf0, bf1;
            #pragma unroll
            for (int v = 0; v < 8; ++v) {
                // A 16x32 bf16 layout: VGPR v holds K = (v<4?2v:2v+8) + 8*half
                int ka = ((v < 4) ? (2 * v) : (2 * v + 8)) + 8 * half + kb;
                af.u[v]  = *(const unsigned*)&Alds[wm * 16 + l15][ka];
                // B 32x16 bf16 layout: VGPR v holds K = (v<4?2v:2v+8) + 8*half, col = l15
                int kbi = ((v < 4) ? (2 * v) : (2 * v + 8)) + 8 * half + kb;
                bf0.u[v] = *(const unsigned*)&Blds[wn * 32 + l15][kbi];
                bf1.u[v] = *(const unsigned*)&Blds[wn * 32 + 16 + l15][kbi];
            }
            acc0 = __builtin_amdgcn_wmma_f32_16x16x32_bf16(false, af.v, false, bf0.v,
                                                           (short)0, acc0, false, false);
            acc1 = __builtin_amdgcn_wmma_f32_16x16x32_bf16(false, af.v, false, bf1.v,
                                                           (short)0, acc1, false, false);
        }
        __syncthreads();
    }

    // C layout: VGPR r -> M = r + 8*half, N = l15
    #pragma unroll
    for (int r = 0; r < 8; ++r) {
        int row = m0 + wm * 16 + 8 * half + r;
        if (row < M) {
            C[(size_t)row * N + n0 + wn * 32 + l15]      = acc0[r];
            C[(size_t)row * N + n0 + wn * 32 + 16 + l15] = acc1[r];
        }
    }
}

// ---------------------------------------------------------------------------
// Layer 1 edge kernels (H=8, C=64, feature width 512)
// ---------------------------------------------------------------------------
__global__ void edge_score1(const float* __restrict__ xl, const float* __restrict__ xr,
                            const float* __restrict__ att,
                            const int* __restrict__ src, const int* __restrict__ dst,
                            float* __restrict__ alpha, unsigned* __restrict__ amaxk) {
    int t = blockIdx.x * blockDim.x + threadIdx.x;
    if (t >= NEDGES * HEADS) return;
    int e = t >> 3, h = t & 7;
    int s = src[e], d = dst[e];
    const float* pl = xl + (size_t)s * HC + h * HID;
    const float* pr = xr + (size_t)d * HC + h * HID;
    const float* pa = att + h * HID;
    float sum = 0.f;
    #pragma unroll 8
    for (int c = 0; c < HID; ++c)
        sum += leaky(pl[c] + pr[c]) * pa[c];
    alpha[t] = sum;
    atomicMax(&amaxk[(size_t)d * HEADS + h], fkey(sum));
}

__global__ void edge_exp1(const int* __restrict__ dst, const unsigned* __restrict__ amaxk,
                          float* __restrict__ alpha, float* __restrict__ denom) {
    int t = blockIdx.x * blockDim.x + threadIdx.x;
    if (t >= NEDGES * HEADS) return;
    int e = t >> 3, h = t & 7;
    int d = dst[e];
    float m = funkey(amaxk[(size_t)d * HEADS + h]);
    if (!(m > -1e38f && m < 1e38f)) m = 0.f;   // mirrors isfinite -> 0
    float ex = __expf(alpha[t] - m);
    alpha[t] = ex;
    atomicAdd(&denom[(size_t)d * HEADS + h], ex);
}

__global__ __launch_bounds__(256)
void edge_aggregate1(const float* __restrict__ xl,
                     const int* __restrict__ src, const int* __restrict__ dst,
                     const float* __restrict__ alpha, const float* __restrict__ denom,
                     float* __restrict__ acc) {
    int e = blockIdx.x;
    int s = src[e], d = dst[e];
    #pragma unroll
    for (int i = 0; i < 2; ++i) {
        int c = threadIdx.x + i * 256;   // 0..511
        int h = c >> 6;
        float w = alpha[(size_t)e * HEADS + h] / (denom[(size_t)d * HEADS + h] + EPS);
        atomicAdd(&acc[(size_t)d * HC + c], xl[(size_t)s * HC + c] * w);
    }
}

__global__ void finalize1(float* __restrict__ acc, const float* __restrict__ b) {
    size_t t = (size_t)blockIdx.x * blockDim.x + threadIdx.x;
    if (t >= (size_t)NNODES * HC) return;
    float v = acc[t] + b[t & (HC - 1)];
    acc[t] = v > 0.f ? v : (__expf(v) - 1.f);   // ELU
}

// ---------------------------------------------------------------------------
// Layer 2 edge kernels (H=1, C=64)
// ---------------------------------------------------------------------------
__global__ void edge_score2(const float* __restrict__ xl, const float* __restrict__ xr,
                            const float* __restrict__ att,
                            const int* __restrict__ src, const int* __restrict__ dst,
                            float* __restrict__ alpha, unsigned* __restrict__ amaxk) {
    int e = blockIdx.x * blockDim.x + threadIdx.x;
    if (e >= NEDGES) return;
    int s = src[e], d = dst[e];
    const float* pl = xl + (size_t)s * NCLS;
    const float* pr = xr + (size_t)d * NCLS;
    float sum = 0.f;
    #pragma unroll 8
    for (int c = 0; c < NCLS; ++c)
        sum += leaky(pl[c] + pr[c]) * att[c];
    alpha[e] = sum;
    atomicMax(&amaxk[d], fkey(sum));
}

__global__ void edge_exp2(const int* __restrict__ dst, const unsigned* __restrict__ amaxk,
                          float* __restrict__ alpha, float* __restrict__ denom) {
    int e = blockIdx.x * blockDim.x + threadIdx.x;
    if (e >= NEDGES) return;
    int d = dst[e];
    float m = funkey(amaxk[d]);
    if (!(m > -1e38f && m < 1e38f)) m = 0.f;
    float ex = __expf(alpha[e] - m);
    alpha[e] = ex;
    atomicAdd(&denom[d], ex);
}

__global__ void edge_aggregate2(const float* __restrict__ xl,
                                const int* __restrict__ src, const int* __restrict__ dst,
                                const float* __restrict__ alpha, const float* __restrict__ denom,
                                float* __restrict__ out) {
    int t = blockIdx.x * blockDim.x + threadIdx.x;
    if (t >= NEDGES * NCLS) return;
    int e = t >> 6, c = t & 63;
    int s = src[e], d = dst[e];
    float w = alpha[e] / (denom[d] + EPS);
    atomicAdd(&out[(size_t)d * NCLS + c], xl[(size_t)s * NCLS + c] * w);
}

__global__ void finalize2(float* __restrict__ out, const float* __restrict__ b) {
    int t = blockIdx.x * blockDim.x + threadIdx.x;
    if (t >= NNODES * NCLS) return;
    out[t] += b[t & (NCLS - 1)];
}

// ---------------------------------------------------------------------------
// Host-side orchestration
// ---------------------------------------------------------------------------
extern "C" void kernel_launch(void* const* d_in, const int* in_sizes, int n_in,
                              void* d_out, int out_size, void* d_ws, size_t ws_size,
                              hipStream_t stream) {
    const float* x    = (const float*)d_in[0];
    const int*   ei   = (const int*)  d_in[1];
    const float* Wl1  = (const float*)d_in[2];
    const float* Wr1  = (const float*)d_in[3];
    const float* att1 = (const float*)d_in[4];
    const float* b1   = (const float*)d_in[5];
    const float* Wl2  = (const float*)d_in[6];
    const float* Wr2  = (const float*)d_in[7];
    const float* att2 = (const float*)d_in[8];
    const float* b2   = (const float*)d_in[9];
    float* out = (float*)d_out;

    const int* src = ei;
    const int* dst = ei + NEDGES;

    // workspace layout (float units)
    float* ws = (float*)d_ws;
    size_t off = 0;
    float*    XL1    = ws + off; off += (size_t)NNODES * HC;      // 20000*512
    float*    XR1    = ws + off; off += (size_t)NNODES * HC;
    float*    ALPHA1 = ws + off; off += (size_t)NEDGES * HEADS;
    unsigned* AMK1   = (unsigned*)(ws + off); off += (size_t)NNODES * HEADS;
    float*    DEN1   = ws + off; off += (size_t)NNODES * HEADS;
    float*    ACC1   = ws + off; off += (size_t)NNODES * HC;      // becomes h after ELU
    float*    XL2    = ws + off; off += (size_t)NNODES * NCLS;
    float*    XR2    = ws + off; off += (size_t)NNODES * NCLS;
    float*    ALPHA2 = ws + off; off += (size_t)NEDGES;
    unsigned* AMK2   = (unsigned*)(ws + off); off += (size_t)NNODES;
    float*    DEN2   = ws + off; off += (size_t)NNODES;
    (void)ws_size; (void)n_in; (void)in_sizes;

    // zero-init accumulators / max-keys (key 0 == "-NaN" sentinel < everything)
    hipMemsetAsync(AMK1, 0, (size_t)NNODES * HEADS * 4, stream);
    hipMemsetAsync(DEN1, 0, (size_t)NNODES * HEADS * 4, stream);
    hipMemsetAsync(ACC1, 0, (size_t)NNODES * HC * 4, stream);
    hipMemsetAsync(AMK2, 0, (size_t)NNODES * 4, stream);
    hipMemsetAsync(DEN2, 0, (size_t)NNODES * 4, stream);
    hipMemsetAsync(out,  0, (size_t)out_size * 4, stream);

    const int mblk = (NNODES + 63) / 64;   // 313

    // ---- layer 1 projections: [20000,256] x [256,512]
    {
        dim3 g(mblk, HC / 64);
        gemm_bf16_wmma<<<g, 256, 0, stream>>>(x, Wl1, XL1, NNODES, HC, F_IN);
        gemm_bf16_wmma<<<g, 256, 0, stream>>>(x, Wr1, XR1, NNODES, HC, F_IN);
    }
    // ---- layer 1 edge softmax + aggregation
    {
        int nt = NEDGES * HEADS;
        edge_score1<<<(nt + 255) / 256, 256, 0, stream>>>(XL1, XR1, att1, src, dst, ALPHA1, AMK1);
        edge_exp1  <<<(nt + 255) / 256, 256, 0, stream>>>(dst, AMK1, ALPHA1, DEN1);
        edge_aggregate1<<<NEDGES, 256, 0, stream>>>(XL1, src, dst, ALPHA1, DEN1, ACC1);
        size_t n = (size_t)NNODES * HC;
        finalize1<<<(unsigned)((n + 255) / 256), 256, 0, stream>>>(ACC1, b1);
    }
    // ---- layer 2 projections: [20000,512] x [512,64]
    {
        dim3 g(mblk, NCLS / 64);
        gemm_bf16_wmma<<<g, 256, 0, stream>>>(ACC1, Wl2, XL2, NNODES, NCLS, HC);
        gemm_bf16_wmma<<<g, 256, 0, stream>>>(ACC1, Wr2, XR2, NNODES, NCLS, HC);
    }
    // ---- layer 2 edge softmax + aggregation into d_out
    {
        edge_score2<<<(NEDGES + 255) / 256, 256, 0, stream>>>(XL2, XR2, att2, src, dst, ALPHA2, AMK2);
        edge_exp2  <<<(NEDGES + 255) / 256, 256, 0, stream>>>(dst, AMK2, ALPHA2, DEN2);
        int nt = NEDGES * NCLS;
        edge_aggregate2<<<(nt + 255) / 256, 256, 0, stream>>>(XL2, src, dst, ALPHA2, DEN2, out);
        finalize2<<<(NNODES * NCLS + 255) / 256, 256, 0, stream>>>(out, b2);
    }
}
